// WavLMMultiheadAttention_90271622627746
// MI455X (gfx1250) — matrix operand order repaired and checked by reading, not compiled, not run on hardware
//
#include <hip/hip_runtime.h>
#include <math.h>
#include <stdint.h>

// ---- problem constants ----
#define E_DIM 1024
#define H_DIM 16
#define T_DIM 2048
#define B_DIM 2
#define HD_DIM 64
#define NTOK (T_DIM * B_DIM)
#define NBUCK 32
#define BIAS_TAB 4096
#define KV_TILE 64                       // keys per attention iteration

typedef __attribute__((ext_vector_type(16))) __bf16 v16bf;
typedef __attribute__((ext_vector_type(8)))  __bf16 v8bf;
typedef __attribute__((ext_vector_type(4)))  __bf16 v4bf;
typedef __attribute__((ext_vector_type(8)))  float  v8f;
typedef __attribute__((ext_vector_type(4)))  int    v4i;
typedef __attribute__((ext_vector_type(8)))  int    v8i;
typedef __attribute__((ext_vector_type(4)))  unsigned v4u;

#if __has_builtin(__builtin_amdgcn_global_load_async_to_lds_b128)
#define HAS_ASYNC_LDS 1
#endif
#if __has_builtin(__builtin_amdgcn_tensor_load_to_lds)
#define HAS_TDM 1
#endif

// pointer-to-vec4i in the global / LDS address spaces (async builtin's expected types)
typedef __attribute__((address_space(1))) v4i GV4;
typedef __attribute__((address_space(3))) v4i LV4;

static __device__ inline v8f zero8() {
    v8f z = {0.f, 0.f, 0.f, 0.f, 0.f, 0.f, 0.f, 0.f};
    return z;
}

static __device__ inline v8f wmma_bf16(v16bf a, v16bf b, v8f c) {
    return __builtin_amdgcn_wmma_f32_16x16x32_bf16(false, a, false, b, (short)0, c, false, false);
}

// A-fragment: 16x32 bf16 (lane&15 = row; lanes 0-15 K={k0..+7, k0+16..+23}, lanes 16-31 the +8 halves)
static __device__ inline v16bf load_frag_a(const __bf16* base, size_t ld, int row0, int k0, int lane) {
    int m = row0 + (lane & 15);
    int half = (lane >> 4) & 1;
    const __bf16* p = base + (size_t)m * ld + (size_t)(k0 + half * 8);
    v8bf lo = *reinterpret_cast<const v8bf*>(p);
    v8bf hi = *reinterpret_cast<const v8bf*>(p + 16);
    return __builtin_shufflevector(lo, hi, 0,1,2,3,4,5,6,7,8,9,10,11,12,13,14,15);
}

// B-fragment: 32x16 bf16 (lane&15 = col; lanes 0-15 K={k0..+7, k0+16..+23}, lanes 16-31 the +8 halves)
static __device__ inline v16bf load_frag_b(const __bf16* base, size_t ldn, int n0, int k0, int lane) {
    int n = n0 + (lane & 15);
    int half = (lane >> 4) & 1;
    const __bf16* p = base + (size_t)n * ldn + (size_t)(k0 + half * 8);
    v8bf lo = *reinterpret_cast<const v8bf*>(p);
    v8bf hi = *reinterpret_cast<const v8bf*>(p + 16);
    return __builtin_shufflevector(lo, hi, 0,1,2,3,4,5,6,7,8,9,10,11,12,13,14,15);
}

// ---- cross-lane reductions over 16-lane halves: DPP8 for xor 1/2/4, bpermute for xor 8 ----
template<int M>
static __device__ inline float xor_dpp8(float v) {
#if __has_builtin(__builtin_amdgcn_mov_dpp8)
    constexpr unsigned sel = (unsigned)((0 ^ M) | ((1 ^ M) << 3) | ((2 ^ M) << 6) | ((3 ^ M) << 9) |
                                        ((4 ^ M) << 12) | ((5 ^ M) << 15) | ((6 ^ M) << 18) | ((7 ^ M) << 21));
    int r = __builtin_amdgcn_mov_dpp8(__builtin_bit_cast(int, v), sel);
    return __builtin_bit_cast(float, r);
#else
    return __shfl_xor(v, M, 32);
#endif
}

static __device__ inline float red_max16(float v) {
    v = fmaxf(v, xor_dpp8<1>(v));
    v = fmaxf(v, xor_dpp8<2>(v));
    v = fmaxf(v, xor_dpp8<4>(v));
    v = fmaxf(v, __shfl_xor(v, 8, 32));
    return v;
}
static __device__ inline float red_sum16(float v) {
    v += xor_dpp8<1>(v);
    v += xor_dpp8<2>(v);
    v += xor_dpp8<4>(v);
    v += __shfl_xor(v, 8, 32);
    return v;
}

// ---- staging: TDM (preferred) / async global->LDS / sync fallback ----
static __device__ inline void async_copy16(const __bf16* g, __bf16* l) {
#ifdef HAS_ASYNC_LDS
    __builtin_amdgcn_global_load_async_to_lds_b128((GV4*)g, (LV4*)l, 0, 0);
#else
    *reinterpret_cast<v8bf*>(l) = *reinterpret_cast<const v8bf*>(g);
#endif
}

// per-thread cooperative copy of a 64-row x 64-bf16 tile (fallback path)
static __device__ inline void stage_tile(const __bf16* gbase, size_t gstride, __bf16* lbase, int tid) {
    #pragma unroll
    for (int i = 0; i < 2; ++i) {
        int idx = tid + i * 256;       // 0..511
        int row = idx >> 3;
        int chunk = (idx & 7) * 8;
        async_copy16(gbase + (size_t)row * gstride + chunk, lbase + row * KV_TILE + chunk);
    }
}

#ifdef HAS_TDM
static __device__ inline unsigned lds_addr_of(const __bf16* l) {
    return (unsigned)(uintptr_t)(LV4*)l;
}

// One TDM descriptor moves a 64x64 bf16 tile (row stride 2048 elems) to LDS.
// D# packing per CDNA5 ISA ch.8: group0 = {count, lds_addr, global_addr, type=2},
// group1 = {data_size=2B, tensor_dim0/1=64, tile_dim0/1=64, dim0_stride=2048}.
static __device__ inline void tdm_load_tile(const __bf16* g, unsigned ldsAddr) {
    unsigned long long ga = (unsigned long long)(uintptr_t)g;
    v4u g0 = { 1u,                                            // count=1, is_restore=0, gather=0
               ldsAddr,                                       // lds_addr[31:0]
               (unsigned)(ga & 0xffffffffu),                  // global_addr[31:0]
               (unsigned)((ga >> 32) & 0x01ffffffu) | 0x80000000u };  // addr[56:32] | type=2<<30
    v8i g1 = { (int)0x00010000u,        // workgroup_mask=0, data_size=1 (2 bytes)
               (int)(64u << 16),        // tensor_dim0[15:0]=64 at bits [63:48]
               (int)(64u << 16),        // tensor_dim0[31:16]=0 | tensor_dim1[15:0]=64
               (int)(64u << 16),        // tensor_dim1[31:16]=0 | tile_dim0=64
               (int)64,                 // tile_dim1=64 | tile_dim2=0
               (int)2048,               // tensor_dim0_stride[31:0] = 2048 elems
               0,                       // stride[47:32]=0 | tensor_dim1_stride[15:0]=0
               0 };                     // tensor_dim1_stride[47:16]=0
    v4i zz4 = {0, 0, 0, 0};
    v8i zz8 = {0, 0, 0, 0, 0, 0, 0, 0};
    __builtin_amdgcn_tensor_load_to_lds(g0, g1, zz4, zz4, zz8, 0);
}
#endif

static __device__ inline void wait_stage() {
#ifdef HAS_TDM
#if __has_builtin(__builtin_amdgcn_s_wait_tensorcnt)
    __builtin_amdgcn_s_wait_tensorcnt(0);
#else
    asm volatile("s_wait_tensorcnt 0" ::: "memory");
#endif
#elif defined(HAS_ASYNC_LDS)
#if __has_builtin(__builtin_amdgcn_s_wait_asynccnt)
    __builtin_amdgcn_s_wait_asynccnt(0);
#else
    asm volatile("s_wait_asynccnt 0" ::: "memory");
#endif
#endif
}

// ---------------------------------------------------------------------------
// f32 -> bf16 conversion (4 elems/thread)
// ---------------------------------------------------------------------------
__global__ __launch_bounds__(256) void cvt_bf16_kernel(const float* __restrict__ src,
                                                       unsigned short* __restrict__ dst, int n) {
    int i = (blockIdx.x * 256 + threadIdx.x) * 4;
    if (i + 3 < n) {
        float4 f = *reinterpret_cast<const float4*>(src + i);
        v4bf o = {(__bf16)f.x, (__bf16)f.y, (__bf16)f.z, (__bf16)f.w};
        *reinterpret_cast<v4bf*>(dst + i) = o;
    }
}

// ---------------------------------------------------------------------------
// Gated relative-position coefficient:  gA1[b,h,t] = ga*(gb*grep_a[h]-1)+2
// ---------------------------------------------------------------------------
__global__ __launch_bounds__(256) void gate_kernel(const float* __restrict__ query,
                                                   const float* __restrict__ grep_w,
                                                   const float* __restrict__ grep_b,
                                                   const float* __restrict__ grep_a,
                                                   float* __restrict__ gA1) {
    int idx = blockIdx.x * 256 + threadIdx.x;
    if (idx >= B_DIM * H_DIM * T_DIM) return;
    int t = idx % T_DIM;
    int h = (idx / T_DIM) % H_DIM;
    int b = idx / (T_DIM * H_DIM);
    const float* x = query + ((size_t)t * B_DIM + b) * E_DIM + h * HD_DIM;
    float ga = 0.f, gb = 0.f;
    #pragma unroll
    for (int j = 0; j < 8; ++j) {
        float d = grep_b[j];
        const float* w = grep_w + j * HD_DIM;
        for (int dd = 0; dd < HD_DIM; ++dd) d += x[dd] * w[dd];
        if (j < 4) ga += d; else gb += d;
    }
    ga = 1.f / (1.f + __expf(-ga));
    gb = 1.f / (1.f + __expf(-gb));
    gA1[((size_t)b * H_DIM + h) * T_DIM + t] = ga * (gb * grep_a[h] - 1.f) + 2.f;
}

// ---------------------------------------------------------------------------
// Relative-bucket bias table:  biasRel[h][rp+2047] = rel_emb[bucket(rp)][h]
// ---------------------------------------------------------------------------
__global__ __launch_bounds__(256) void relbias_kernel(const float* __restrict__ rel_emb,
                                                      float* __restrict__ biasRel) {
    int idx = blockIdx.x * 256 + threadIdx.x;
    if (idx >= H_DIM * BIAS_TAB) return;
    int h = idx >> 12;
    int i = idx & (BIAS_TAB - 1);
    int rp = i - (T_DIM - 1);
    int bucket = (rp > 0) ? (NBUCK / 2) : 0;
    int rpa = rp < 0 ? -rp : rp;
    int add;
    if (rpa < 8) {
        add = rpa;
    } else {
        float rpaf = (float)(rpa < 1 ? 1 : rpa);
        int lg = 8 + (int)(__logf(rpaf / 8.f) / __logf(16.f) * 8.f);
        add = lg < 15 ? lg : 15;
    }
    biasRel[idx] = rel_emb[(bucket + add) * H_DIM + h];
}

// ---------------------------------------------------------------------------
// WMMA GEMM:  D = (A(MxK) @ W(NxK)^T + bias) * scale
// MODE 0: bf16 row-major   MODE 1: bf16 V-transposed [b][h][d][t]   MODE 2: f32
// ---------------------------------------------------------------------------
template<int MODE>
static __device__ inline void gemm_store_tile(const v8f& acc, int m0, int n0,
                                              const float* __restrict__ bias, float scale, int N,
                                              __bf16* __restrict__ outb, float* __restrict__ outf,
                                              int lane) {
    int half = lane >> 4, col = lane & 15;
    #pragma unroll
    for (int r = 0; r < 8; ++r) {
        int m = m0 + r + 8 * half;
        int n = n0 + col;
        float v = (acc[r] + bias[n]) * scale;
        if (MODE == 2) {
            outf[(size_t)m * N + n] = v;
        } else if (MODE == 0) {
            outb[(size_t)m * N + n] = (__bf16)v;
        } else {
            int t = m >> 1, bb = m & 1;
            int hh = n >> 6, d = n & 63;
            outb[(((size_t)bb * H_DIM + hh) * HD_DIM + d) * T_DIM + t] = (__bf16)v;
        }
    }
}

template<int MODE>
__global__ __launch_bounds__(256) void gemm_bf16_kernel(const unsigned short* __restrict__ Araw,
                                                        const unsigned short* __restrict__ Wraw,
                                                        const float* __restrict__ bias, float scale,
                                                        int M, int N, int K,
                                                        unsigned short* __restrict__ outraw,
                                                        float* __restrict__ outf) {
    const __bf16* A = reinterpret_cast<const __bf16*>(Araw);
    const __bf16* W = reinterpret_cast<const __bf16*>(Wraw);
    __bf16* outb = reinterpret_cast<__bf16*>(outraw);

    int lane = threadIdx.x & 31;
    int wid = threadIdx.x >> 5;
    int r0 = blockIdx.y * 128 + (wid >> 1) * 32;
    int c0 = blockIdx.x * 64 + (wid & 1) * 32;

    v8f acc00 = zero8(), acc01 = zero8(), acc10 = zero8(), acc11 = zero8();
    for (int k0 = 0; k0 < K; k0 += 32) {
        v16bf a0 = load_frag_a(A, (size_t)K, r0, k0, lane);
        v16bf a1 = load_frag_a(A, (size_t)K, r0 + 16, k0, lane);
        v16bf b0 = load_frag_b(W, (size_t)K, c0, k0, lane);
        v16bf b1 = load_frag_b(W, (size_t)K, c0 + 16, k0, lane);
        acc00 = wmma_bf16(a0, b0, acc00);
        acc01 = wmma_bf16(a0, b1, acc01);
        acc10 = wmma_bf16(a1, b0, acc10);
        acc11 = wmma_bf16(a1, b1, acc11);
    }
    gemm_store_tile<MODE>(acc00, r0,      c0,      bias, scale, N, outb, outf, lane);
    gemm_store_tile<MODE>(acc01, r0,      c0 + 16, bias, scale, N, outb, outf, lane);
    gemm_store_tile<MODE>(acc10, r0 + 16, c0,      bias, scale, N, outb, outf, lane);
    gemm_store_tile<MODE>(acc11, r0 + 16, c0 + 16, bias, scale, N, outb, outf, lane);
}

// ---------------------------------------------------------------------------
// Flash attention per (b,h): K/V tiles (64 keys) staged in LDS by the Tensor
// Data Mover (double buffered, TENSORcnt-tracked); each wave owns 16 query
// rows, online softmax with gated relative-position bias.
// grid = (T/128, H, B), block = 256 (8 waves).
// ---------------------------------------------------------------------------
__global__ __launch_bounds__(256) void attn_kernel(const unsigned short* __restrict__ qraw,
                                                   const unsigned short* __restrict__ kraw,
                                                   const unsigned short* __restrict__ vraw,
                                                   const float* __restrict__ gA1,
                                                   const float* __restrict__ biasRel,
                                                   unsigned short* __restrict__ ctxraw) {
    const __bf16* q  = reinterpret_cast<const __bf16*>(qraw);
    const __bf16* k  = reinterpret_cast<const __bf16*>(kraw);
    const __bf16* vT = reinterpret_cast<const __bf16*>(vraw);
    __bf16* ctx = reinterpret_cast<__bf16*>(ctxraw);

    __shared__ float sBias[BIAS_TAB];                         // 16 KB
    __shared__ __align__(32) __bf16 sK[2][KV_TILE * HD_DIM];  // 2 x 8 KB, [s][d]
    __shared__ __align__(32) __bf16 sV[2][HD_DIM * KV_TILE];  // 2 x 8 KB, [d][s]
    __shared__ __align__(32) __bf16 sP[8][16 * KV_TILE];      // 16 KB, per-wave P tile

    const int h = blockIdx.y, b = blockIdx.z;
    const int tid = threadIdx.x;
    for (int i = tid; i < BIAS_TAB; i += 256) sBias[i] = biasRel[h * BIAS_TAB + i];

    const int lane = tid & 31;
    const int wid = tid >> 5;
    const int half = lane >> 4;
    const int col = lane & 15;
    const int t0 = blockIdx.x * 128 + wid * 16;

    const __bf16* qbase = q + (size_t)b * E_DIM + h * HD_DIM;             // row stride B*E
    const __bf16* kbase = k + (size_t)b * E_DIM + h * HD_DIM;
    const __bf16* vbase = vT + ((size_t)b * H_DIM + h) * HD_DIM * T_DIM;  // [d][t]

    v16bf qa0 = load_frag_a(qbase, (size_t)B_DIM * E_DIM, t0, 0, lane);
    v16bf qa1 = load_frag_a(qbase, (size_t)B_DIM * E_DIM, t0, 32, lane);

    float mrow[8], lrow[8], gr[8];
    v8f acc0 = zero8(), acc1 = zero8(), acc2 = zero8(), acc3 = zero8();
    #pragma unroll
    for (int r = 0; r < 8; ++r) {
        mrow[r] = -3.0e38f;
        lrow[r] = 0.f;
        gr[r] = gA1[((size_t)b * H_DIM + h) * T_DIM + t0 + r + 8 * half];
    }

    // prologue: stage tile 0 into buffer 0
#ifdef HAS_TDM
    if (wid == 0) {
        tdm_load_tile(kbase, lds_addr_of(sK[0]));
        tdm_load_tile(vbase, lds_addr_of(sV[0]));
    }
#else
    stage_tile(kbase, (size_t)B_DIM * E_DIM, sK[0], tid);
    stage_tile(vbase, (size_t)T_DIM, sV[0], tid);
#endif

    const int NIT = T_DIM / KV_TILE;                     // 32
    for (int it = 0; it < NIT; ++it) {
        wait_stage();             // staged tile committed to LDS (TENSORcnt/ASYNCcnt)
        __syncthreads();          // everyone's writes visible; prior reads retired
        const int cur = it & 1;
        if (it + 1 < NIT) {       // overlap: issue next tile while computing
            int s0n = (it + 1) * KV_TILE;
#ifdef HAS_TDM
            if (wid == 0) {
                tdm_load_tile(kbase + (size_t)s0n * B_DIM * E_DIM, lds_addr_of(sK[cur ^ 1]));
                tdm_load_tile(vbase + s0n, lds_addr_of(sV[cur ^ 1]));
            }
#else
            stage_tile(kbase + (size_t)s0n * B_DIM * E_DIM, (size_t)B_DIM * E_DIM, sK[cur ^ 1], tid);
            stage_tile(vbase + s0n, (size_t)T_DIM, sV[cur ^ 1], tid);
#endif
        }
        const __bf16* Kt = sK[cur];
        const __bf16* Vt = sV[cur];

        // S = Q @ K^T for 64 key columns (four 16x16 C tiles)
        v8f S[4];
        #pragma unroll
        for (int sc = 0; sc < 4; ++sc) {
            S[sc] = wmma_bf16(qa0, load_frag_b(Kt, (size_t)HD_DIM, sc * 16, 0, lane), zero8());
            S[sc] = wmma_bf16(qa1, load_frag_b(Kt, (size_t)HD_DIM, sc * 16, 32, lane), S[sc]);
        }

        // bias + online softmax; P written to wave-private LDS in A-layout order
        const int sbase = it * KV_TILE + col - (t0 + 8 * half) + (T_DIM - 1);
        #pragma unroll
        for (int r = 0; r < 8; ++r) {
            float e0 = S[0][r] + gr[r] * sBias[sbase - r];
            float e1 = S[1][r] + gr[r] * sBias[sbase - r + 16];
            float e2 = S[2][r] + gr[r] * sBias[sbase - r + 32];
            float e3 = S[3][r] + gr[r] * sBias[sbase - r + 48];
            float mx = red_max16(fmaxf(fmaxf(e0, e1), fmaxf(e2, e3)));
            float mnew = fmaxf(mrow[r], mx);
            float cs = __expf(mrow[r] - mnew);
            float p0 = __expf(e0 - mnew);
            float p1 = __expf(e1 - mnew);
            float p2 = __expf(e2 - mnew);
            float p3 = __expf(e3 - mnew);
            float rs = red_sum16((p0 + p1) + (p2 + p3));
            lrow[r] = lrow[r] * cs + rs;
            mrow[r] = mnew;
            acc0[r] *= cs; acc1[r] *= cs; acc2[r] *= cs; acc3[r] *= cs;
            __bf16* pw = &sP[wid][(r + 8 * half) * KV_TILE + col];
            pw[0]  = (__bf16)p0;
            pw[16] = (__bf16)p1;
            pw[32] = (__bf16)p2;
            pw[48] = (__bf16)p3;
        }

        // reload P as A-fragments (C-layout -> A-layout transpose via wave-private LDS)
        v16bf pa0 = load_frag_a(&sP[wid][0], (size_t)KV_TILE, 0, 0, lane);
        v16bf pa1 = load_frag_a(&sP[wid][0], (size_t)KV_TILE, 0, 32, lane);

        // O += P(16x64) @ V(64x64): V tile is [d][s] so K-dim (s) is contiguous
        acc0 = wmma_bf16(pa0, load_frag_b(Vt, (size_t)KV_TILE, 0,  0,  lane), acc0);
        acc1 = wmma_bf16(pa0, load_frag_b(Vt, (size_t)KV_TILE, 16, 0,  lane), acc1);
        acc2 = wmma_bf16(pa0, load_frag_b(Vt, (size_t)KV_TILE, 32, 0,  lane), acc2);
        acc3 = wmma_bf16(pa0, load_frag_b(Vt, (size_t)KV_TILE, 48, 0,  lane), acc3);
        acc0 = wmma_bf16(pa1, load_frag_b(Vt, (size_t)KV_TILE, 0,  32, lane), acc0);
        acc1 = wmma_bf16(pa1, load_frag_b(Vt, (size_t)KV_TILE, 16, 32, lane), acc1);
        acc2 = wmma_bf16(pa1, load_frag_b(Vt, (size_t)KV_TILE, 32, 32, lane), acc2);
        acc3 = wmma_bf16(pa1, load_frag_b(Vt, (size_t)KV_TILE, 48, 32, lane), acc3);
    }

    // normalize and write context (bf16, (t,b,e) row-major) for the out-proj GEMM
    #pragma unroll
    for (int r = 0; r < 8; ++r) {
        int t = t0 + r + 8 * half;
        float inv = 1.0f / lrow[r];
        size_t base = ((size_t)t * B_DIM + b) * E_DIM + h * HD_DIM + col;
        ctx[base + 0]  = (__bf16)(acc0[r] * inv);
        ctx[base + 16] = (__bf16)(acc1[r] * inv);
        ctx[base + 32] = (__bf16)(acc2[r] * inv);
        ctx[base + 48] = (__bf16)(acc3[r] * inv);
    }
}

// ---------------------------------------------------------------------------
extern "C" void kernel_launch(void* const* d_in, const int* in_sizes, int n_in,
                              void* d_out, int out_size, void* d_ws, size_t ws_size,
                              hipStream_t stream) {
    (void)in_sizes; (void)n_in; (void)out_size; (void)ws_size;
    const float* query   = (const float*)d_in[0];
    const float* q_w     = (const float*)d_in[1];
    const float* q_b     = (const float*)d_in[2];
    const float* k_w     = (const float*)d_in[3];
    const float* k_b     = (const float*)d_in[4];
    const float* v_w     = (const float*)d_in[5];
    const float* v_b     = (const float*)d_in[6];
    const float* out_w   = (const float*)d_in[7];
    const float* out_b   = (const float*)d_in[8];
    const float* rel_emb = (const float*)d_in[9];
    const float* grep_w  = (const float*)d_in[10];
    const float* grep_b  = (const float*)d_in[11];
    const float* grep_a  = (const float*)d_in[12];

    unsigned short* xbuf = (unsigned short*)d_ws;                       // 4096*1024
    unsigned short* wq   = xbuf + (size_t)NTOK * E_DIM;
    unsigned short* wk   = wq + (size_t)E_DIM * E_DIM;
    unsigned short* wv   = wk + (size_t)E_DIM * E_DIM;
    unsigned short* wo   = wv + (size_t)E_DIM * E_DIM;
    unsigned short* qbuf = wo + (size_t)E_DIM * E_DIM;
    unsigned short* kbuf = qbuf + (size_t)NTOK * E_DIM;
    unsigned short* vT   = kbuf + (size_t)NTOK * E_DIM;
    unsigned short* ctx  = vT + (size_t)NTOK * E_DIM;
    float* gA1     = (float*)(ctx + (size_t)NTOK * E_DIM);
    float* biasRel = gA1 + (size_t)B_DIM * H_DIM * T_DIM;

    const int nX = NTOK * E_DIM;
    const int nW = E_DIM * E_DIM;

    cvt_bf16_kernel<<<nX / 1024, 256, 0, stream>>>(query, xbuf, nX);
    cvt_bf16_kernel<<<nW / 1024, 256, 0, stream>>>(q_w, wq, nW);
    cvt_bf16_kernel<<<nW / 1024, 256, 0, stream>>>(k_w, wk, nW);
    cvt_bf16_kernel<<<nW / 1024, 256, 0, stream>>>(v_w, wv, nW);
    cvt_bf16_kernel<<<nW / 1024, 256, 0, stream>>>(out_w, wo, nW);

    gate_kernel<<<(B_DIM * H_DIM * T_DIM) / 256, 256, 0, stream>>>(query, grep_w, grep_b, grep_a, gA1);
    relbias_kernel<<<(H_DIM * BIAS_TAB) / 256, 256, 0, stream>>>(rel_emb, biasRel);

    dim3 gg(E_DIM / 64, NTOK / 128);
    const float scaling = 0.125f;   // HD^-0.5
    gemm_bf16_kernel<0><<<gg, 256, 0, stream>>>(xbuf, wq, q_b, scaling, NTOK, E_DIM, E_DIM, qbuf, nullptr);
    gemm_bf16_kernel<0><<<gg, 256, 0, stream>>>(xbuf, wk, k_b, 1.0f,    NTOK, E_DIM, E_DIM, kbuf, nullptr);
    gemm_bf16_kernel<1><<<gg, 256, 0, stream>>>(xbuf, wv, v_b, 1.0f,    NTOK, E_DIM, E_DIM, vT,   nullptr);

    attn_kernel<<<dim3(T_DIM / 128, H_DIM, B_DIM), 256, 0, stream>>>(qbuf, kbuf, vT, gA1, biasRel, ctx);

    gemm_bf16_kernel<2><<<gg, 256, 0, stream>>>(ctx, wo, out_b, 1.0f, NTOK, E_DIM, E_DIM, nullptr, (float*)d_out);
}
